// TimeEncoder_72069551227407
// MI455X (gfx1250) — matrix operand order repaired and checked by reading, not compiled, not run on hardware
//
#include <hip/hip_runtime.h>
#include <hip/hip_bf16.h>

// ---------------------------------------------------------------------------
// Fused 2-layer LSTM + dense head for MI455X (gfx1250, wave32, WMMA).
// One persistent block per 16-row batch tile; all weights resident in LDS.
// Gate nonlinearities use the CDNA5 hardware V_TANH_F32 transcendental.
// ---------------------------------------------------------------------------

typedef __attribute__((ext_vector_type(16))) __bf16 v16bf;
typedef __attribute__((ext_vector_type(8)))  __bf16 v8bf;
typedef __attribute__((ext_vector_type(8)))  float  v8f;

#define T_STEPS 128
#define NB      1024     // batch
#define H1      64
#define H2      128
#define DOUT    128
// padded LDS row strides (bf16 elems) to avoid bank conflicts on A-loads
#define H1S     72
#define H2S     136

// Hardware tanh (single TRANS op on gfx1250); fall back to libm if absent.
#if __has_builtin(__builtin_amdgcn_tanhf)
__device__ __forceinline__ float htanh(float x) { return __builtin_amdgcn_tanhf(x); }
#elif __has_builtin(__builtin_amdgcn_tanh_f32)
__device__ __forceinline__ float htanh(float x) { return __builtin_amdgcn_tanh_f32(x); }
#else
__device__ __forceinline__ float htanh(float x) { return tanhf(x); }
#endif

// sigmoid(x) = 0.5*tanh(x/2) + 0.5  -> one TRANS + one FMA, branchless
__device__ __forceinline__ float sigf(float x) {
  return __builtin_fmaf(0.5f, htanh(0.5f * x), 0.5f);
}

// swizzled weight blob index: per 32x16 (KxN) WMMA B tile, 32 lanes x 16 bf16
// contiguous per lane.  KT = K>>5 (number of 32-row k-tiles).
__device__ __forceinline__ int swz(int k, int n, int KT) {
  int tile = (n >> 4) * KT + (k >> 5);
  int ln   = (((k >> 4) & 1) << 4) | (n & 15);
  return (tile << 9) | (ln << 4) | (k & 15);
}

// WMMA B operand: one contiguous 32B read per lane from swizzled blob.
__device__ __forceinline__ v16bf ldB(const __bf16* w, int kt, int nt, int KT, int lane) {
  return *(const v16bf*)(w + (((nt * KT + kt) << 9) + (lane << 4)));
}

// WMMA A operand from row-major [16 x K] bf16 LDS buffer (padded stride).
// lane holds m = lane%16; contiguous K-run [b16, b16+16) of the
// 32-wide chunk kc  (matches swizzled B: k = 16*(lane/16) + s).
__device__ __forceinline__ v16bf ldA(const __bf16* h, int stride, int kc, int lane) {
  int m   = lane & 15;
  int b16 = (lane >> 4) << 4;
  const __bf16* p = h + m * stride + (kc << 5) + b16;
  v8bf lo = *(const v8bf*)p;
  v8bf hi = *(const v8bf*)(p + 8);
  return __builtin_shufflevector(lo, hi, 0, 1, 2, 3, 4, 5, 6, 7,
                                         8, 9, 10, 11, 12, 13, 14, 15);
}

__device__ __forceinline__ v8f wmma_bf16(v16bf a, v16bf b, v8f c) {
  return __builtin_amdgcn_wmma_f32_16x16x32_bf16(false, a, false, b,
                                                 (short)0, c, false, false);
}

__global__ __launch_bounds__(256, 1)
void lstm2_fused(const int*   __restrict__ x,
                 const float* __restrict__ E,
                 const float* __restrict__ W1,
                 const float* __restrict__ U1,
                 const float* __restrict__ b1,
                 const float* __restrict__ W2,
                 const float* __restrict__ U2,
                 const float* __restrict__ b2,
                 const float* __restrict__ Wd,
                 const float* __restrict__ bd,
                 float*       __restrict__ out)
{
  // ---- LDS residency (~310 KB of the 320 KB WGP budget) ----
  __shared__ __align__(64) __bf16 Gt [18 * 6 * 256];   //  55,296 B  x@W1+b1 LUT
  __shared__ __align__(64) __bf16 U1s[64 * 256];       //  32,768 B
  __shared__ __align__(64) __bf16 W2s[64 * 512];       //  65,536 B
  __shared__ __align__(64) __bf16 U2s[128 * 512];      // 131,072 B
  __shared__ float  b2s[512];                          //   2,048 B
  __shared__ float  zx [16 * 256];                     //  16,384 B  per-step x@W1 tile
  __shared__ __align__(64) __bf16 h1b[16 * H1S];       //   2,304 B
  __shared__ __align__(64) __bf16 h2b[16 * H2S];       //   4,352 B
  __shared__ int    xb[96];                            //     384 B

  const int tid  = threadIdx.x;
  const int lane = tid & 31;
  const int wv   = tid >> 5;        // 8 waves
  const int nl   = lane & 15;
  const int mh   = (lane >> 4) << 3; // +8 row offset for upper half-wave
  const int b0   = blockIdx.x << 4;  // 16-row batch tile

  // ================= Phase A: build LDS-resident operands =================
  // G[v][f][n] = E[v,:] . W1[f*32:(f+1)*32, n]  (+ b1 folded into f==0)
  for (int e = tid; e < 18 * 6 * 256; e += 256) {
    int v = e / 1536, rem = e - v * 1536;
    int f = rem >> 8, n = rem & 255;
    float acc = (f == 0) ? b1[n] : 0.0f;
    const float* Er = E + v * 32;
    const float* Wc = W1 + (f * 32) * 256 + n;
#pragma unroll 8
    for (int k = 0; k < 32; ++k) acc += Er[k] * Wc[k * 256];
    Gt[e] = (__bf16)acc;
  }
  for (int e = tid; e < 64 * 256; e += 256) {          // U1: 64 x 256
    int k = e >> 8, n = e & 255;
    U1s[swz(k, n, 2)] = (__bf16)U1[e];
  }
  for (int e = tid; e < 64 * 512; e += 256) {          // W2: 64 x 512
    int k = e >> 9, n = e & 511;
    W2s[swz(k, n, 2)] = (__bf16)W2[e];
  }
  for (int e = tid; e < 128 * 512; e += 256) {         // U2: 128 x 512
    int k = e >> 9, n = e & 511;
    U2s[swz(k, n, 4)] = (__bf16)U2[e];
  }
  for (int e = tid; e < 512; e += 256) b2s[e] = b2[e];
  for (int e = tid; e < 16 * H1S; e += 256) h1b[e] = (__bf16)0.0f;
  for (int e = tid; e < 16 * H2S; e += 256) h2b[e] = (__bf16)0.0f;

  float c1r[8], c2r[8];
#pragma unroll
  for (int r = 0; r < 8; ++r) { c1r[r] = 0.0f; c2r[r] = 0.0f; }

  __syncthreads();

  // ============================ time loop ============================
  for (int t = 0; t < T_STEPS; ++t) {
    // -- x indices for this step (16 rows x 6 fields) --
    if (tid < 96) {
      int row = tid / 6, f = tid - row * 6;
      xb[tid] = x[(b0 + row) * (T_STEPS * 6) + t * 6 + f];
    }
    __syncthreads();                                   // B0: xb ready

    // -- zx[m][n] = sum_f G[x[m,f]][f][n]  (all 8 waves) --
    for (int e = tid; e < 4096; e += 256) {
      int m = e >> 8, n = e & 255;
      const int* xr = xb + m * 6;
      float s = 0.0f;
#pragma unroll
      for (int f = 0; f < 6; ++f)
        s += (float)Gt[(xr[f] * 6 + f) * 256 + n];
      zx[e] = s;
    }
    __syncthreads();                                   // B1: zx ready

    // -- layer 1: z1 = zx + h1_prev @ U1; waves 0..3 own 16 hidden each --
    v16bf a10, a11;
    if (wv < 4) {
      a10 = ldA(h1b, H1S, 0, lane);
      a11 = ldA(h1b, H1S, 1, lane);
    }
    __syncthreads();                                   // B2: h1 reads done

    if (wv < 4) {
      v8f zt[4];
#pragma unroll
      for (int g = 0; g < 4; ++g) {
        int nt   = g * 4 + wv;
        int col0 = g * 64 + (wv << 4);
        v8f acc;
#pragma unroll
        for (int r = 0; r < 8; ++r) acc[r] = zx[(r + mh) * 256 + col0 + nl];
        acc = wmma_bf16(a10, ldB(U1s, 0, nt, 2, lane), acc);
        acc = wmma_bf16(a11, ldB(U1s, 1, nt, 2, lane), acc);
        zt[g] = acc;
      }
#pragma unroll
      for (int r = 0; r < 8; ++r) {
        float ig = sigf(zt[0][r]);
        float fg = sigf(zt[1][r]);
        float gg = htanh(zt[2][r]);
        float og = sigf(zt[3][r]);
        c1r[r] = fg * c1r[r] + ig * gg;
        float h = og * htanh(c1r[r]);
        h1b[(r + mh) * H1S + (wv << 4) + nl] = (__bf16)h;
      }
    }
    __syncthreads();                                   // B3: new h1 visible

    // -- layer 2: z2 = h1 @ W2 + h2_prev @ U2 + b2; all 8 waves --
    v16bf a1n0 = ldA(h1b, H1S, 0, lane);
    v16bf a1n1 = ldA(h1b, H1S, 1, lane);
    v16bf a2[4];
#pragma unroll
    for (int kt = 0; kt < 4; ++kt) a2[kt] = ldA(h2b, H2S, kt, lane);
    __syncthreads();                                   // B4: h2 reads done

    v8f z2[4];
#pragma unroll
    for (int g = 0; g < 4; ++g) {
      int nt   = g * 8 + wv;
      int col0 = g * 128 + (wv << 4);
      v8f acc;
      float bb = b2s[col0 + nl];
#pragma unroll
      for (int r = 0; r < 8; ++r) acc[r] = bb;
      acc = wmma_bf16(a1n0, ldB(W2s, 0, nt, 2, lane), acc);
      acc = wmma_bf16(a1n1, ldB(W2s, 1, nt, 2, lane), acc);
#pragma unroll
      for (int kt = 0; kt < 4; ++kt)
        acc = wmma_bf16(a2[kt], ldB(U2s, kt, nt, 4, lane), acc);
      z2[g] = acc;
    }
#pragma unroll
    for (int r = 0; r < 8; ++r) {
      float ig = sigf(z2[0][r]);
      float fg = sigf(z2[1][r]);
      float gg = htanh(z2[2][r]);
      float og = sigf(z2[3][r]);
      c2r[r] = fg * c2r[r] + ig * gg;
      float h = og * htanh(c2r[r]);
      h2b[(r + mh) * H2S + (wv << 4) + nl] = (__bf16)h;
    }
    // end of step; B0 of next iteration fences h2b writes
  }
  __syncthreads();

  // ================== dense head: relu(h2_T @ Wd + bd) ==================
  {
    v16bf a2[4];
#pragma unroll
    for (int kt = 0; kt < 4; ++kt) a2[kt] = ldA(h2b, H2S, kt, lane);

    int col = (wv << 4) + nl;
    v8f acc;
    float bb = bd[col];
#pragma unroll
    for (int r = 0; r < 8; ++r) acc[r] = bb;

#pragma unroll
    for (int kt = 0; kt < 4; ++kt) {
      int kb = kt * 32 + ((lane >> 4) << 4);
      v16bf wb;
#pragma unroll
      for (int j = 0; j < 16; ++j) wb[j] = (__bf16)Wd[(kb + j) * DOUT + col];
      acc = wmma_bf16(a2[kt], wb, acc);
    }
#pragma unroll
    for (int r = 0; r < 8; ++r)
      out[(b0 + r + mh) * DOUT + col] = fmaxf(acc[r], 0.0f);
  }
}

extern "C" void kernel_launch(void* const* d_in, const int* in_sizes, int n_in,
                              void* d_out, int out_size, void* d_ws, size_t ws_size,
                              hipStream_t stream) {
  (void)in_sizes; (void)n_in; (void)out_size; (void)d_ws; (void)ws_size;
  const int*   x  = (const int*)  d_in[0];
  const float* E  = (const float*)d_in[1];
  const float* W1 = (const float*)d_in[2];
  const float* U1 = (const float*)d_in[3];
  const float* b1 = (const float*)d_in[4];
  const float* W2 = (const float*)d_in[5];
  const float* U2 = (const float*)d_in[6];
  const float* b2 = (const float*)d_in[7];
  const float* Wd = (const float*)d_in[8];
  const float* bd = (const float*)d_in[9];
  float* out = (float*)d_out;

  lstm2_fused<<<dim3(NB / 16), dim3(256), 0, stream>>>(
      x, E, W1, U1, b1, W2, U2, b2, Wd, bd, out);
}
